// CausalSelfAttention_27066883899567
// MI455X (gfx1250) — matrix operand order repaired and checked by reading, not compiled, not run on hardware
//
#include <hip/hip_runtime.h>

typedef __bf16 bf16;
typedef __attribute__((ext_vector_type(8)))  bf16  v8bf;
typedef __attribute__((ext_vector_type(16))) bf16  v16bf;
typedef __attribute__((ext_vector_type(8)))  float v8f;
typedef __attribute__((ext_vector_type(4)))  unsigned int v4u;
typedef __attribute__((ext_vector_type(8)))  int v8i_;
typedef __attribute__((ext_vector_type(4)))  int v4i_;

#define B_  2
#define L_  2048
#define E_  2048
#define H_  16
#define HD_ 128
#define E3_ (3 * E_)

#define SCALE_ 0.08838834764831845f   // 1/sqrt(128)
#define LOG2E_ 1.4426950408889634f

union FragAB { v16bf v; v8bf h[2]; };

__device__ __forceinline__ v8f wmma_bf16(const FragAB& a, const FragAB& b, v8f c) {
  return __builtin_amdgcn_wmma_f32_16x16x32_bf16(false, a.v, false, b.v,
                                                 (short)0, c, false, false);
}

// ------------------------------------------------------------- TDM descriptor
// 2-D tile load: tile_d0 contiguous elements (bf16) x tile_d1 rows, row stride
// stride0 (elements). Groups per CDNA5 ISA.
// This toolchain exposes the 6-arg builtin:
//   (uint32x4 g0, int32x8 g1, int32x4 g2, int32x4 g3, int32x8, i32 cpol)
__device__ __forceinline__ void tdm_load_2d(unsigned lds_off, const bf16* gptr,
                                            unsigned tile_d0, unsigned tile_d1,
                                            unsigned tensor_d0, unsigned tensor_d1,
                                            unsigned stride0) {
  unsigned long long ga = (unsigned long long)(uintptr_t)gptr;
  v4u g0 = { 1u,                                     // count=1, user descriptor
             lds_off,                                // lds_addr (bytes)
             (unsigned)ga,                           // global_addr[31:0]
             ((unsigned)(ga >> 32) & 0x1FFFFFFu) | (2u << 30) };  // addr hi | type=2
  v8i_ g1 = { (int)(1u << 16),                                   // data_size = 2B
              (int)((tensor_d0 & 0xFFFFu) << 16),                // tensor_dim0 lo
              (int)((tensor_d0 >> 16) | ((tensor_d1 & 0xFFFFu) << 16)),
              (int)((tensor_d1 >> 16) | (tile_d0 << 16)),        // tile_dim0
              (int)(tile_d1 & 0xFFFFu),                          // tile_dim1, dim2=0
              (int)stride0,                                      // dim0_stride lo
              0, 0 };
  v4i_ z4 = {0, 0, 0, 0};
  v8i_ z8 = {0, 0, 0, 0, 0, 0, 0, 0};
  __builtin_amdgcn_tensor_load_to_lds(g0, g1, z4, z4, z8, 0);
}

// ---------------------------------------------------------------- converts
__global__ __launch_bounds__(256)
void k_f32_to_bf16(const float* __restrict__ in, bf16* __restrict__ out, int n) {
  int i = blockIdx.x * 256 + threadIdx.x;
  int stride = gridDim.x * 256;
  for (; i < n; i += stride) out[i] = (bf16)in[i];
}

// ---------------------------------------------------------------- GEMM (TDM)
// C[M,N] = A[M,K] * Bw[N,K]^T   (A, Bw bf16 row-major, K contiguous)
// 256 thr = 8 waves; block tile 128x128; wave tile 32x64.
// K staged in 64-wide chunks via TDM into double-buffered LDS.
template <bool BF16_OUT>
__global__ __launch_bounds__(256)
void k_gemm(const bf16* __restrict__ A, const bf16* __restrict__ Bw,
            void* __restrict__ Cout, int M, int N, int K) {
  __shared__ bf16 tA[2][128 * 64];
  __shared__ bf16 tB[2][128 * 64];

  const int tid  = threadIdx.x;
  const int w    = tid >> 5;
  const int lane = tid & 31;
  const int g    = lane >> 4;     // half-wave
  const int lm   = lane & 15;
  const int am   = (w & 3) * 32;  // wave row offset within block tile
  const int bn   = (w >> 2) * 64; // wave col offset within block tile
  const int m0   = blockIdx.y * 128;
  const int n0   = blockIdx.x * 128;
  const int NCH  = K >> 6;        // number of 64-wide K chunks

  v8f zero = {};
  v8f acc[2][4];
#pragma unroll
  for (int i = 0; i < 2; i++)
#pragma unroll
    for (int t = 0; t < 4; t++) acc[i][t] = zero;

  if (w == 0) {  // one TDM issue per workgroup (EXEC ignored by TDM)
    tdm_load_2d((unsigned)(uintptr_t)&tA[0][0], A + (size_t)m0 * K,
                64, 128, (unsigned)K, 1u << 20, (unsigned)K);
    tdm_load_2d((unsigned)(uintptr_t)&tB[0][0], Bw + (size_t)n0 * K,
                64, 128, (unsigned)K, 1u << 20, (unsigned)K);
  }

  for (int c = 0; c < NCH; ++c) {
    const int cur = c & 1;
    if (w == 0) {
      if (c + 1 < NCH) {  // prefetch next chunk, then retire current (in-order)
        const int nxt = cur ^ 1;
        const size_t ko = (size_t)(c + 1) * 64;
        tdm_load_2d((unsigned)(uintptr_t)&tA[nxt][0], A + (size_t)m0 * K + ko,
                    64, 128, (unsigned)K, 1u << 20, (unsigned)K);
        tdm_load_2d((unsigned)(uintptr_t)&tB[nxt][0], Bw + (size_t)n0 * K + ko,
                    64, 128, (unsigned)K, 1u << 20, (unsigned)K);
        __builtin_amdgcn_s_wait_tensorcnt(2);
      } else {
        __builtin_amdgcn_s_wait_tensorcnt(0);
      }
    }
    __syncthreads();  // chunk c visible to all waves

    const bf16* la0 = &tA[cur][(am + lm) * 64];
    const bf16* la1 = &tA[cur][(am + 16 + lm) * 64];
#pragma unroll
    for (int kk = 0; kk < 64; kk += 32) {
      FragAB a0, a1;
      a0.h[0] = *(const v8bf*)(la0 + kk + 8 * g);
      a0.h[1] = *(const v8bf*)(la0 + kk + 16 + 8 * g);
      a1.h[0] = *(const v8bf*)(la1 + kk + 8 * g);
      a1.h[1] = *(const v8bf*)(la1 + kk + 16 + 8 * g);
#pragma unroll
      for (int t = 0; t < 4; t++) {
        const bf16* lb = &tB[cur][(bn + 16 * t + lm) * 64];
        FragAB bt;
        bt.h[0] = *(const v8bf*)(lb + kk + 8 * g);
        bt.h[1] = *(const v8bf*)(lb + kk + 16 + 8 * g);
        acc[0][t] = wmma_bf16(a0, bt, acc[0][t]);
        acc[1][t] = wmma_bf16(a1, bt, acc[1][t]);
      }
    }
    __syncthreads();  // all waves done with buffer before TDM overwrites it
  }

#pragma unroll
  for (int i = 0; i < 2; i++)
#pragma unroll
    for (int t = 0; t < 4; t++)
#pragma unroll
      for (int r = 0; r < 8; r++) {
        int row = m0 + am + 16 * i + r + 8 * g;
        int col = n0 + bn + 16 * t + lm;
        float v = acc[i][t][r];
        if (BF16_OUT) ((bf16*)Cout)[(size_t)row * N + col] = (bf16)v;
        else          ((float*)Cout)[(size_t)row * N + col] = v;
      }
}

// ---------------------------------------------------------------- RoPE + repack
// qkv bf16 [B,L,3E] -> Qh,Kh bf16 [B,H,L,HD] (rope applied), Vt bf16 [B,H,HD,L]
__global__ __launch_bounds__(256)
void k_rope_pack(const bf16* __restrict__ qkv, const float* __restrict__ rope,
                 bf16* __restrict__ Qh, bf16* __restrict__ Kh, bf16* __restrict__ Vt) {
  int idx = blockIdx.x * 256 + threadIdx.x;   // B*L*H*(HD/2) = 4194304 threads
  int j = idx & 63;
  int h = (idx >> 6) & 15;
  int l = (idx >> 10) & 2047;
  int b = idx >> 21;

  float sn = rope[((size_t)l * 64 + j) * 2 + 0];
  float cs = rope[((size_t)l * 64 + j) * 2 + 1];

  const bf16* base = qkv + ((size_t)b * L_ + l) * E3_;
  size_t hoff = (size_t)h * HD_ + 2 * j;
  size_t qo = (((size_t)(b * H_ + h)) * L_ + l) * HD_ + 2 * j;

  float a0 = (float)base[hoff];
  float a1 = (float)base[hoff + 1];
  Qh[qo]     = (bf16)(a0 * cs - a1 * sn);
  Qh[qo + 1] = (bf16)(a1 * cs + a0 * sn);

  a0 = (float)base[E_ + hoff];
  a1 = (float)base[E_ + hoff + 1];
  Kh[qo]     = (bf16)(a0 * cs - a1 * sn);
  Kh[qo + 1] = (bf16)(a1 * cs + a0 * sn);

  a0 = (float)base[2 * E_ + hoff];
  a1 = (float)base[2 * E_ + hoff + 1];
  size_t vo = (((size_t)(b * H_ + h)) * HD_ + 2 * j) * L_ + l;
  Vt[vo]      = (bf16)a0;
  Vt[vo + L_] = (bf16)a1;
}

// ---------------------------------------------------------------- flash attention
// grid (L/128, B*H); 8 waves/block, each wave owns a 16-row q tile.
__global__ __launch_bounds__(256)
void k_flash(const bf16* __restrict__ Qh, const bf16* __restrict__ Kh,
             const bf16* __restrict__ Vt, bf16* __restrict__ Y) {
  __shared__ bf16 pbuf[8 * 16 * 32];

  const int tid  = threadIdx.x;
  const int w    = tid >> 5;
  const int lane = tid & 31;
  const int g    = lane >> 4;
  const int lm   = lane & 15;
  const int bh   = blockIdx.y;       // b*H + h
  const int b    = bh >> 4;
  const int h    = bh & 15;
  const int q0   = blockIdx.x * 128 + w * 16;

  bf16* pb = pbuf + w * (16 * 32);

  // preload Q fragments for the wave's 16 q rows (4 chunks of K=32 over HD)
  const bf16* qrow = Qh + ((size_t)bh * L_ + q0 + lm) * HD_;
  FragAB qf[4];
#pragma unroll
  for (int dc = 0; dc < 4; dc++) {
    qf[dc].h[0] = *(const v8bf*)(qrow + dc * 32 + 8 * g);
    qf[dc].h[1] = *(const v8bf*)(qrow + dc * 32 + 16 + 8 * g);
  }

  v8f zero = {};
  float mrow[8], lrow[8];
  v8f o[8];
#pragma unroll
  for (int r = 0; r < 8; r++) { mrow[r] = -3.0e38f; lrow[r] = 0.0f; }
#pragma unroll
  for (int t = 0; t < 8; t++) o[t] = zero;

  const bf16* kbase = Kh + (size_t)bh * L_ * HD_;
  const bf16* vbase = Vt + (size_t)bh * HD_ * L_;

  for (int k0 = 0; k0 <= q0 + 15; k0 += 32) {
    // ---- scores S = Q K^T for 32 keys (two 16x16 tiles)
    v8f s0 = zero, s1 = zero;
    const bf16* krow0 = kbase + (size_t)(k0 + lm) * HD_;
    const bf16* krow1 = kbase + (size_t)(k0 + 16 + lm) * HD_;
#pragma unroll
    for (int dc = 0; dc < 4; dc++) {
      FragAB kb;
      kb.h[0] = *(const v8bf*)(krow0 + dc * 32 + 8 * g);
      kb.h[1] = *(const v8bf*)(krow0 + dc * 32 + 16 + 8 * g);
      s0 = wmma_bf16(qf[dc], kb, s0);
      kb.h[0] = *(const v8bf*)(krow1 + dc * 32 + 8 * g);
      kb.h[1] = *(const v8bf*)(krow1 + dc * 32 + 16 + 8 * g);
      s1 = wmma_bf16(qf[dc], kb, s1);
    }

    // ---- online softmax (row = (r, half); 16 lanes hold the 16 key cols)
    float alpha[8];
#pragma unroll
    for (int r = 0; r < 8; r++) {
      int q = q0 + r + 8 * g;
      float x0 = s0[r] * SCALE_;
      float x1 = s1[r] * SCALE_;
      if (k0 + lm > q)      x0 = -3.0e38f;
      if (k0 + 16 + lm > q) x1 = -3.0e38f;
      float mx = fmaxf(x0, x1);
#pragma unroll
      for (int off = 1; off < 16; off <<= 1) mx = fmaxf(mx, __shfl_xor(mx, off, 16));
      float mnew = fmaxf(mrow[r], mx);
      float p0 = __builtin_amdgcn_exp2f((x0 - mnew) * LOG2E_);
      float p1 = __builtin_amdgcn_exp2f((x1 - mnew) * LOG2E_);
      float rs = p0 + p1;
#pragma unroll
      for (int off = 1; off < 16; off <<= 1) rs += __shfl_xor(rs, off, 16);
      float al = __builtin_amdgcn_exp2f((mrow[r] - mnew) * LOG2E_);
      lrow[r] = lrow[r] * al + rs;
      mrow[r] = mnew;
      alpha[r] = al;
      // P tile to LDS in row-major [16 x 32] for A-fragment reload
      pb[(r + 8 * g) * 32 + lm]      = (bf16)p0;
      pb[(r + 8 * g) * 32 + 16 + lm] = (bf16)p1;
    }

    // rescale accumulator
#pragma unroll
    for (int t = 0; t < 8; t++)
#pragma unroll
      for (int r = 0; r < 8; r++) o[t][r] *= alpha[r];

    // wave-local LDS store->load ordering (cross-lane within the wave)
    __asm volatile("s_wait_dscnt 0" ::: "memory");

    FragAB pf;
    pf.h[0] = *(const v8bf*)(pb + lm * 32 + 8 * g);
    pf.h[1] = *(const v8bf*)(pb + lm * 32 + 16 + 8 * g);

    // ---- O += P * V  (V transposed: rows = d, contiguous along keys)
#pragma unroll
    for (int t = 0; t < 8; t++) {
      const bf16* vrow = vbase + (size_t)(t * 16 + lm) * L_ + k0 + 8 * g;
      FragAB vb;
      vb.h[0] = *(const v8bf*)(vrow);
      vb.h[1] = *(const v8bf*)(vrow + 16);
      o[t] = wmma_bf16(pf, vb, o[t]);
    }
  }

  float inv[8];
#pragma unroll
  for (int r = 0; r < 8; r++) inv[r] = 1.0f / lrow[r];

#pragma unroll
  for (int t = 0; t < 8; t++)
#pragma unroll
    for (int r = 0; r < 8; r++) {
      int row = q0 + r + 8 * g;
      Y[((size_t)b * L_ + row) * E_ + (size_t)h * HD_ + t * 16 + lm] =
          (bf16)(o[t][r] * inv[r]);
    }
}

// ---------------------------------------------------------------- launch
extern "C" void kernel_launch(void* const* d_in, const int* in_sizes, int n_in,
                              void* d_out, int out_size, void* d_ws, size_t ws_size,
                              hipStream_t stream) {
  const float* x      = (const float*)d_in[0];
  const float* rope   = (const float*)d_in[1];
  const float* w_attn = (const float*)d_in[2];
  const float* w_proj = (const float*)d_in[3];

  const size_t n_x   = (size_t)B_ * L_ * E_;       // 8,388,608
  const size_t n_wa  = (size_t)3 * E_ * E_;        // 12,582,912
  const size_t n_wp  = (size_t)E_ * E_;            // 4,194,304
  const size_t n_qkv = (size_t)B_ * L_ * 3 * E_;   // 25,165,824
  const size_t n_hd  = (size_t)B_ * H_ * L_ * HD_; // 8,388,608

  bf16* ws   = (bf16*)d_ws;                 // ~160 MiB of bf16 staging
  bf16* xb   = ws;
  bf16* wab  = xb + n_x;
  bf16* wpb  = wab + n_wa;
  bf16* qkvb = wpb + n_wp;
  bf16* Qh   = qkvb + n_qkv;
  bf16* Kh   = Qh + n_hd;
  bf16* Vt   = Kh + n_hd;
  bf16* Yb   = Vt + n_hd;

  k_f32_to_bf16<<<2048, 256, 0, stream>>>(x, xb, (int)n_x);
  k_f32_to_bf16<<<2048, 256, 0, stream>>>(w_attn, wab, (int)n_wa);
  k_f32_to_bf16<<<2048, 256, 0, stream>>>(w_proj, wpb, (int)n_wp);

  // qkv = x @ w_attn^T : M=4096, N=6144, K=2048  (bf16 out)
  k_gemm<true><<<dim3(6144 / 128, 4096 / 128), 256, 0, stream>>>(
      xb, wab, (void*)qkvb, 4096, 6144, 2048);

  k_rope_pack<<<(B_ * L_ * H_ * 64) / 256, 256, 0, stream>>>(qkvb, rope, Qh, Kh, Vt);

  k_flash<<<dim3(L_ / 128, B_ * H_), 256, 0, stream>>>(Qh, Kh, Vt, Yb);

  // out = y @ w_proj^T : M=4096, N=2048, K=2048  (fp32 out)
  k_gemm<false><<<dim3(2048 / 128, 4096 / 128), 256, 0, stream>>>(
      Yb, wpb, d_out, 4096, 2048, 2048);
  (void)in_sizes; (void)n_in; (void)out_size; (void)ws_size;
}
